// HierarchicalMemoryWorker_32392643346608
// MI455X (gfx1250) — compile-verified, matrix operand order checked by reading, not run on hardware
//
#include <hip/hip_runtime.h>
#include <hip/hip_bf16.h>
#include <stdint.h>

// ---------------- problem constants ----------------
constexpr int kB  = 2;
constexpr int kS  = 1024;
constexpr int kBS = kB * kS;      // 2048 tokens
constexpr int kE  = 512;
constexpr int kN  = 8;            // sectors
constexpr int kM  = 2048;         // memories per sector
constexpr int kKD = 32;
constexpr int kVD = 32;
constexpr int kK  = 4;            // top-k
constexpr float kLnEps = 1e-5f;

constexpr int kParts   = 8;                        // M partitions for top-k GEMM
constexpr int kMemsPerPart = kM / kParts;          // 256
constexpr int kTilesPerPart = kMemsPerPart / 16;   // 16 m-tiles per block
constexpr int kTokBlk  = 128;                      // tokens per block (8 waves x 16)
constexpr int kWaves   = kTokBlk / 16;             // 8 waves
constexpr int kRowPitch = kE * 2 + 16;             // 1040 B: bank-conflict-free LDS rows
constexpr int kKChunks = kE / 32;                  // 16 WMMA K-steps

// ---------------- WMMA types ----------------
typedef __attribute__((ext_vector_type(16))) __bf16 v16bf;
typedef __attribute__((ext_vector_type(8)))  float  v8f;

union V16U {
  v16bf v;
  uint4 q[2];
  unsigned short u[16];
};

__device__ __forceinline__ unsigned short f32_to_bf16_rne(float f) {
  union { float f; unsigned int u; } c; c.f = f;
  unsigned int u = c.u;
  unsigned int lsb = (u >> 16) & 1u;
  u += 0x7fffu + lsb;
  return (unsigned short)(u >> 16);
}

// ---------------- kernel 1: memory_keys -> bf16 ----------------
__global__ __launch_bounds__(256) void convert_keys_kernel(
    const float* __restrict__ mk, unsigned short* __restrict__ mkb, int n) {
  int i = blockIdx.x * blockDim.x + threadIdx.x;
  if (i < n) mkb[i] = f32_to_bf16_rne(mk[i]);
}

// ---------------- kernel 2: gather embeddings (f32 + bf16) ----------------
__global__ __launch_bounds__(256) void gather_embed_kernel(
    const int* __restrict__ x, const float* __restrict__ emb,
    float* __restrict__ xe, unsigned short* __restrict__ xeb) {
  int t = blockIdx.x;
  int row = x[t];
  const float* src = emb + (size_t)row * kE;
  for (int e = threadIdx.x; e < kE; e += blockDim.x) {
    float v = src[e];
    xe [(size_t)t * kE + e] = v;
    xeb[(size_t)t * kE + e] = f32_to_bf16_rne(v);
  }
}

// ---------------- kernel 3: sector scores + softmax ----------------
__global__ __launch_bounds__(32) void sector_dist_kernel(
    const float* __restrict__ xe, const float* __restrict__ skeys,
    float* __restrict__ sdist) {
  int t = blockIdx.x;
  int lane = threadIdx.x;
  const float* xr = xe + (size_t)t * kE;
  float acc[kN];
#pragma unroll
  for (int n = 0; n < kN; ++n) acc[n] = 0.f;
  for (int e = lane; e < kE; e += 32) {
    float xv = xr[e];
#pragma unroll
    for (int n = 0; n < kN; ++n) acc[n] += xv * skeys[n * kE + e];
  }
#pragma unroll
  for (int n = 0; n < kN; ++n)
    for (int off = 16; off > 0; off >>= 1)
      acc[n] += __shfl_xor(acc[n], off, 32);
  if (lane == 0) {
    float mx = acc[0];
#pragma unroll
    for (int n = 1; n < kN; ++n) mx = fmaxf(mx, acc[n]);
    float ex[kN]; float s = 0.f;
#pragma unroll
    for (int n = 0; n < kN; ++n) { ex[n] = expf(acc[n] - mx); s += ex[n]; }
    float inv = 1.0f / s;
#pragma unroll
    for (int n = 0; n < kN; ++n) sdist[(size_t)t * kN + n] = ex[n] * inv;
  }
}

// ---------------- kernel 4: token query = x_emb @ Wq + bq ----------------
__global__ __launch_bounds__(32) void token_query_kernel(
    const float* __restrict__ xe, const float* __restrict__ Wq,
    const float* __restrict__ bq, float* __restrict__ tq) {
  int t = blockIdx.x;
  int d = threadIdx.x;                 // 0..31
  const float* xr = xe + (size_t)t * kE;
  float acc = bq[d];
  for (int e = 0; e < kE; ++e) acc += xr[e] * Wq[e * kKD + d];
  tq[(size_t)t * kKD + d] = acc;
}

// ---------------- kernel 5: WMMA bf16 score GEMM + streaming top-4 ----------------
// grid = (BS/128, N, PARTS), block = 256 (8 waves). B tiles (16 mems x K=512,
// 16KB) are double-buffered into LDS with GLOBAL_LOAD_ASYNC_TO_LDS_B128 and
// shared by all 8 waves. Each wave's 16-token A panel is loaded ONCE into a
// 128-VGPR register array; B fragments are double-buffered in registers so the
// next ds_load_b128 pair flows under the current v_wmma (partial dscnt waits).

__device__ __forceinline__ void async_load_tile(
    uint64_t gbase, int m0, const char* lds_buf, int tid) {
#pragma unroll
  for (int i = 0; i < 4; ++i) {
    int c   = tid + i * 256;              // 0..1023 : 16B chunk id
    int r   = c >> 6;                     // mem row 0..15
    int col = c & 63;                     // 16B chunk within row
    unsigned int goff = (unsigned int)(((m0 + r) * kE + col * 8) * 2);
    unsigned int loff = (unsigned int)(uintptr_t)(lds_buf + r * kRowPitch + col * 16);
    asm volatile("global_load_async_to_lds_b128 %0, %1, %2"
                 :: "v"(loff), "v"(goff), "s"(gbase)
                 : "memory");
  }
}

__global__ __launch_bounds__(256) void topk_scores_kernel(
    const unsigned short* __restrict__ xeb, const unsigned short* __restrict__ mkb,
    float* __restrict__ pval, int* __restrict__ pidx) {
  __shared__ __align__(16) char s_b[2][16 * kRowPitch];   // 33.3 KB double buffer
  __shared__ float s_sc[kWaves][16][17];                  // per-wave score tiles

  const int tid    = threadIdx.x;
  const int wave   = tid >> 5;
  const int lane   = tid & 31;
  const int t0     = blockIdx.x * kTokBlk + wave * 16;
  const int sector = blockIdx.y;
  const int part   = blockIdx.z;
  const int rowm   = lane & 15;
  const int kb     = (lane < 16) ? 0 : 8;   // ISA 16-bit A/B fragment K-base

  const unsigned short* Abase = xeb + (size_t)(t0 + rowm) * kE + kb;
  const uint64_t gsec = (uint64_t)(uintptr_t)(mkb + (size_t)sector * kM * kE);
  const int mbase = part * kMemsPerPart;

  // prefetch first B tile, then pin the A panel in registers (loop-invariant)
  async_load_tile(gsec, mbase, s_b[0], tid);

  V16U af[kKChunks];
#pragma unroll
  for (int q = 0; q < kKChunks; ++q) {
    const unsigned short* ap = Abase + q * 32;
    af[q].q[0] = *(const uint4*)(ap);        // K kb..kb+7
    af[q].q[1] = *(const uint4*)(ap + 16);   // K kb+16..kb+23
  }

  float tv0 = -1e30f, tv1 = -1e30f, tv2 = -1e30f, tv3 = -1e30f;
  int   ti0 = 0, ti1 = 0, ti2 = 0, ti3 = 0;

  for (int mt = 0; mt < kTilesPerPart; ++mt) {
    const int m0 = mbase + mt * 16;

    if (mt + 1 < kTilesPerPart) {
      async_load_tile(gsec, m0 + 16, s_b[(mt + 1) & 1], tid);
      asm volatile("s_wait_asynccnt 0x4" ::: "memory");   // tile mt landed
    } else {
      asm volatile("s_wait_asynccnt 0x0" ::: "memory");
    }
    __syncthreads();   // all waves' chunks of buf[mt&1] visible; s_sc free

    // B fragment base for this lane: its column row in the staged tile
    const char* lrow = &s_b[mt & 1][(size_t)rowm * kRowPitch + 2 * kb];

    // register double-buffer for B fragments
    V16U bfr[2];
    bfr[0].q[0] = *(const uint4*)(lrow);
    bfr[0].q[1] = *(const uint4*)(lrow + 32);

    v8f c = {0.f, 0.f, 0.f, 0.f, 0.f, 0.f, 0.f, 0.f};
#pragma unroll
    for (int q = 0; q < kKChunks; ++q) {
      const int nq = q + 1;
      if (nq < kKChunks) {
        bfr[nq & 1].q[0] = *(const uint4*)(lrow + 64 * nq);
        bfr[nq & 1].q[1] = *(const uint4*)(lrow + 64 * nq + 32);
      }
      c = __builtin_amdgcn_wmma_f32_16x16x32_bf16(
          false, af[q].v, false, bfr[q & 1].v, (short)0, c, false, false);
    }

    // C/D layout: lane<16 -> col=lane rows 0..7 ; lane>=16 -> col=lane-16 rows 8..15
    const int col  = lane & 15;
    const int rofs = (lane < 16) ? 0 : 8;
#pragma unroll
    for (int r = 0; r < 8; ++r) s_sc[wave][rofs + r][col] = c[r];
    __syncthreads();   // score tile visible; all waves done reading buf[mt&1]

    if (lane < 16) {
#pragma unroll
      for (int j = 0; j < 16; ++j) {
        float v = s_sc[wave][lane][j];
        int gi = m0 + j;
        if (v > tv3) {
          if (v > tv0)      { tv3=tv2; ti3=ti2; tv2=tv1; ti2=ti1; tv1=tv0; ti1=ti0; tv0=v; ti0=gi; }
          else if (v > tv1) { tv3=tv2; ti3=ti2; tv2=tv1; ti2=ti1; tv1=v;  ti1=gi; }
          else if (v > tv2) { tv3=tv2; ti3=ti2; tv2=v;  ti2=gi; }
          else              { tv3=v;  ti3=gi; }
        }
      }
    }
  }

  if (lane < 16) {
    int t = t0 + lane;
    size_t o = (((size_t)t * kN + sector) * kParts + part) * 4;
    pval[o+0] = tv0; pval[o+1] = tv1; pval[o+2] = tv2; pval[o+3] = tv3;
    pidx[o+0] = ti0; pidx[o+1] = ti1; pidx[o+2] = ti2; pidx[o+3] = ti3;
  }
}

// ---------------- kernel 6: merge partial top-4, softmax weights over top-4 ----------------
__global__ __launch_bounds__(256) void topk_combine_kernel(
    const float* __restrict__ pval, const int* __restrict__ pidx,
    float* __restrict__ topw, int* __restrict__ topidx) {
  int tn = blockIdx.x * blockDim.x + threadIdx.x;
  if (tn >= kBS * kN) return;
  const float* pv = pval + (size_t)tn * kParts * 4;
  const int*   pi = pidx + (size_t)tn * kParts * 4;
  float v[kParts * 4]; int id[kParts * 4];
#pragma unroll
  for (int i = 0; i < kParts * 4; ++i) { v[i] = pv[i]; id[i] = pi[i]; }
  float sv[kK]; int si[kK];
#pragma unroll
  for (int k = 0; k < kK; ++k) {
    int best = 0; float bv = -3e30f; int bi = 0x7fffffff;
#pragma unroll
    for (int i = 0; i < kParts * 4; ++i) {
      bool take = (v[i] > bv) || (v[i] == bv && id[i] < bi);
      if (take) { bv = v[i]; bi = id[i]; best = i; }
    }
    sv[k] = bv; si[k] = bi; v[best] = -3e30f;
  }
  float mx = sv[0];
  float e[kK]; float s = 0.f;
#pragma unroll
  for (int k = 0; k < kK; ++k) { e[k] = expf(sv[k] - mx); s += e[k]; }
  float inv = 1.0f / s;
  size_t o = (size_t)tn * kK;
#pragma unroll
  for (int k = 0; k < kK; ++k) { topw[o + k] = e[k] * inv; topidx[o + k] = si[k]; }
}

// ---------------- kernel 7: ctx gather-GEMM + sector mix + Wo + residual + LN ----------------
__global__ __launch_bounds__(256) void tail_kernel(
    const float* __restrict__ xe, const float* __restrict__ tq,
    const float* __restrict__ know, const float* __restrict__ topw,
    const int* __restrict__ topidx, const float* __restrict__ sdist,
    const float* __restrict__ Wo, const float* __restrict__ bo,
    const float* __restrict__ gamma, const float* __restrict__ beta,
    float* __restrict__ out) {
  __shared__ float s_q[kKD];
  __shared__ float s_ctx[kN][kVD];
  __shared__ float s_final[kVD];
  __shared__ float s_red[256];

  int t = blockIdx.x;
  int tid = threadIdx.x;

  if (tid < kKD) s_q[tid] = tq[(size_t)t * kKD + tid];
  __syncthreads();

  {
    int v = tid & 31;
    int n = tid >> 5;
    float acc = 0.f;
    size_t base_tn = ((size_t)t * kN + n) * kK;
#pragma unroll
    for (int k = 0; k < kK; ++k) {
      int idx  = topidx[base_tn + k];
      float w  = topw[base_tn + k];
      const float* Kt = know + (((size_t)n * kM + idx) * kKD) * kVD;
      float d_acc = 0.f;
#pragma unroll 8
      for (int d = 0; d < kKD; ++d) d_acc += s_q[d] * Kt[d * kVD + v];
      acc += w * d_acc;
    }
    acc *= sdist[(size_t)t * kN + n];
    s_ctx[n][v] = acc;
  }
  __syncthreads();

  if (tid < kVD) {
    float f = 0.f;
#pragma unroll
    for (int n = 0; n < kN; ++n) f += s_ctx[n][tid];
    s_final[tid] = f;
  }
  __syncthreads();

  float h[2];
#pragma unroll
  for (int r = 0; r < 2; ++r) {
    int e = tid + r * 256;
    float p = bo[e];
#pragma unroll 8
    for (int d = 0; d < kVD; ++d) p += s_final[d] * Wo[d * kE + e];
    h[r] = xe[(size_t)t * kE + e] + p;
  }

  s_red[tid] = h[0] + h[1];
  __syncthreads();
  for (int s = 128; s > 0; s >>= 1) {
    if (tid < s) s_red[tid] += s_red[tid + s];
    __syncthreads();
  }
  float mu = s_red[0] * (1.0f / (float)kE);
  __syncthreads();
  float d0 = h[0] - mu, d1 = h[1] - mu;
  s_red[tid] = d0 * d0 + d1 * d1;
  __syncthreads();
  for (int s = 128; s > 0; s >>= 1) {
    if (tid < s) s_red[tid] += s_red[tid + s];
    __syncthreads();
  }
  float var  = s_red[0] * (1.0f / (float)kE);
  float rstd = rsqrtf(var + kLnEps);
#pragma unroll
  for (int r = 0; r < 2; ++r) {
    int e = tid + r * 256;
    out[(size_t)t * kE + e] = (h[r] - mu) * rstd * gamma[e] + beta[e];
  }
}

// ---------------- host launcher ----------------
extern "C" void kernel_launch(void* const* d_in, const int* in_sizes, int n_in,
                              void* d_out, int out_size, void* d_ws, size_t ws_size,
                              hipStream_t stream) {
  const int*   x     = (const int*)  d_in[0];
  // d_in[1] = top_k (fixed 4)
  const float* emb   = (const float*)d_in[2];
  const float* skeys = (const float*)d_in[3];
  const float* mkeys = (const float*)d_in[4];
  const float* know  = (const float*)d_in[5];
  const float* Wq    = (const float*)d_in[6];
  const float* bq    = (const float*)d_in[7];
  const float* Wo    = (const float*)d_in[8];
  const float* bo    = (const float*)d_in[9];
  const float* gamma = (const float*)d_in[10];
  const float* beta  = (const float*)d_in[11];

  float* out_main  = (float*)d_out;                         // (B,S,E)
  float* out_sdist = (float*)d_out + (size_t)kBS * kE;      // (B,S,N)

  // workspace layout (bytes)
  char* ws = (char*)d_ws;
  float*          xe   = (float*)         (ws);                       // 4 MB
  unsigned short* xeb  = (unsigned short*)(ws + 4194304);             // 2 MB
  unsigned short* mkb  = (unsigned short*)(ws + 6291456);             // 16 MB
  float*          tq   = (float*)         (ws + 23068672);            // 256 KB
  float*          pval = (float*)         (ws + 23330816);            // 2 MB
  int*            pidx = (int*)           (ws + 25427968);            // 2 MB
  float*          topw = (float*)         (ws + 27525120);            // 256 KB
  int*            topi = (int*)           (ws + 27787264);            // 256 KB

  const int nMK = kN * kM * kE;  // 8.4M elements
  convert_keys_kernel<<<(nMK + 255) / 256, 256, 0, stream>>>(mkeys, mkb, nMK);
  gather_embed_kernel<<<kBS, 256, 0, stream>>>(x, emb, xe, xeb);
  sector_dist_kernel<<<kBS, 32, 0, stream>>>(xe, skeys, out_sdist);
  token_query_kernel<<<kBS, 32, 0, stream>>>(xe, Wq, bq, tq);
  topk_scores_kernel<<<dim3(kBS / kTokBlk, kN, kParts), 256, 0, stream>>>(xeb, mkb, pval, pidx);
  topk_combine_kernel<<<(kBS * kN + 255) / 256, 256, 0, stream>>>(pval, pidx, topw, topi);
  tail_kernel<<<kBS, 256, 0, stream>>>(xe, tq, know, topw, topi, out_sdist,
                                       Wo, bo, gamma, beta, out_main);
}
